// Head_21827023798611
// MI455X (gfx1250) — matrix operand order repaired and checked by reading, not compiled, not run on hardware
//
#include <hip/hip_runtime.h>
#include <hip/hip_bf16.h>

// Problem constants (from reference): x[8,2048,1024] fp32, W*[64,1024] fp32.
#define B_  8
#define T_  2048
#define E_  1024
#define H_  64
#define BT_ (B_ * T_)

typedef __attribute__((ext_vector_type(16))) __bf16 v16bf;
typedef __attribute__((ext_vector_type(8)))  float  v8f;

// ---------------------------------------------------------------------------
// Kernel 1: q/k/v projections. out = x @ W^T, stored bf16.
//   q, k stored row-major [B*T, 64]; v stored transposed [B][64][T] so that
//   the P·V WMMA B-fragments are contiguous loads later.
// Block: 128 threads = 4 waves; wave w computes the 16-col N-tile w.
// Grid: (BT/16 row tiles, 3 matrices).
// ---------------------------------------------------------------------------
__global__ __launch_bounds__(128) void qkv_proj_kernel(
    const float* __restrict__ x,
    const float* __restrict__ Wq, const float* __restrict__ Wk,
    const float* __restrict__ Wv,
    __bf16* __restrict__ qb, __bf16* __restrict__ kb, __bf16* __restrict__ vTb)
{
    const int wave = threadIdx.x >> 5;
    const int lane = threadIdx.x & 31;
    const int half = lane >> 4;   // which 16-lane half of the wave
    const int l16  = lane & 15;
    const int rowTile = blockIdx.x;        // 16-row tile over B*T
    const int mat     = blockIdx.y;        // 0=q, 1=k, 2=v
    const float* W = (mat == 0) ? Wq : (mat == 1) ? Wk : Wv;

    const size_t arow = (size_t)(rowTile * 16 + l16) * E_;   // A row (x)
    const size_t brow = (size_t)(wave * 16 + l16) * E_;      // B col (W row)

    v8f acc = {};
    for (int k0 = 0; k0 < E_; k0 += 32) {
        // A-frag 16x32 bf16: lane half selects K sub-pattern (two runs of 8)
        v16bf a, b;
        const float* xa = x + arow + k0 + half * 8;
        const float* wb = W + brow + k0 + half * 8;
#pragma unroll
        for (int j = 0; j < 8; ++j) {
            a[j]     = (__bf16)xa[j];
            a[8 + j] = (__bf16)xa[16 + j];
        }
        // B-frag 32x16 bf16: lane holds col l16, same two runs of 8 K as A
#pragma unroll
        for (int j = 0; j < 8; ++j) { b[j] = (__bf16)wb[j]; b[8 + j] = (__bf16)wb[16 + j]; }

        acc = __builtin_amdgcn_wmma_f32_16x16x32_bf16(
            false, a, false, b, (short)0, acc, false, false);
    }

    // C layout: VGPR r -> row (r + 8*half), col l16
    if (mat < 2) {
        __bf16* out = (mat == 0) ? qb : kb;
#pragma unroll
        for (int r = 0; r < 8; ++r)
            out[(size_t)(rowTile * 16 + r + 8 * half) * H_ + wave * 16 + l16] =
                (__bf16)acc[r];
    } else {
        // v transposed: vT[b][h][t]
#pragma unroll
        for (int r = 0; r < 8; ++r) {
            const int gr = rowTile * 16 + r + 8 * half;   // global row = b*T + t
            const int bb = gr >> 11;                      // / T_
            const int t  = gr & (T_ - 1);
            vTb[((size_t)bb * H_ + wave * 16 + l16) * T_ + t] = (__bf16)acc[r];
        }
    }
}

// ---------------------------------------------------------------------------
// Kernel 2: causal flash attention, bf16 WMMA + fp32 online softmax.
// Block: 256 threads = 8 independent waves; each wave owns 16 query rows.
// ---------------------------------------------------------------------------
__device__ __forceinline__ float rmax16(float v)
{
#pragma unroll
    for (int m = 1; m < 16; m <<= 1) v = fmaxf(v, __shfl_xor(v, m, 32));
    return v;
}
__device__ __forceinline__ float rsum16(float v)
{
#pragma unroll
    for (int m = 1; m < 16; m <<= 1) v += __shfl_xor(v, m, 32);
    return v;
}

__global__ __launch_bounds__(256) void attn_kernel(
    const __bf16* __restrict__ qg, const __bf16* __restrict__ kg,
    const __bf16* __restrict__ vTg, float* __restrict__ out)
{
    __shared__ __bf16 pstage[8 * 16 * 64];   // per-wave 16x64 bf16 P staging

    const int wave = threadIdx.x >> 5;
    const int lane = threadIdx.x & 31;
    const int half = lane >> 4;
    const int l16  = lane & 15;

    const int wt = blockIdx.x * 8 + wave;    // wave task 0..1023
    const int b  = wt >> 7;                  // 128 query tiles per batch
    const int q0 = (wt & 127) * 16;

    const __bf16* qb = qg  + (size_t)b * T_ * H_;
    const __bf16* kb = kg  + (size_t)b * T_ * H_;
    const __bf16* vT = vTg + (size_t)b * H_ * T_;
    __bf16* ps = pstage + wave * 16 * 64;

    // Q A-fragments for the 16x64 query tile (two 16x32 frags), loaded once.
    v16bf qf[2];
#pragma unroll
    for (int kk = 0; kk < 2; ++kk) {
        const __bf16* p = qb + (size_t)(q0 + l16) * H_ + kk * 32 + half * 16;
#pragma unroll
        for (int j = 0; j < 16; ++j) qf[kk][j] = p[j];
    }

    float mi[8], li[8];
    v8f o[4] = {};
#pragma unroll
    for (int r = 0; r < 8; ++r) { mi[r] = -1e30f; li[r] = 0.0f; }

    const int ktMax = (q0 + 15) >> 6;        // last 64-key tile touching diag
    for (int kt = 0; kt <= ktMax; ++kt) {
        const int k0 = kt << 6;

        // ---- S = (Q K^T) * scale ; K^T B-frags are contiguous in k rows ----
        v8f s[4];
#pragma unroll
        for (int n = 0; n < 4; ++n) {
            const __bf16* kp = kb + (size_t)(k0 + n * 16 + l16) * H_ + half * 16;
            const v16bf b0 = *(const v16bf*)(kp);        // h = 0..31 half
            const v16bf b1 = *(const v16bf*)(kp + 32);   // h = 32..63 half
            v8f acc = {};
            acc = __builtin_amdgcn_wmma_f32_16x16x32_bf16(
                false, qf[0], false, b0, (short)0, acc, false, false);
            acc = __builtin_amdgcn_wmma_f32_16x16x32_bf16(
                false, qf[1], false, b1, (short)0, acc, false, false);
            s[n] = acc;
        }
        if (kt < ktMax)   // hint next K tile toward the caches
            __builtin_prefetch(kb + (size_t)(k0 + 64 + (lane << 1)) * H_, 0, 0);

        // ---- causal mask + 1/sqrt(64) scale ----
#pragma unroll
        for (int n = 0; n < 4; ++n) {
            const int kcol = k0 + n * 16 + l16;
#pragma unroll
            for (int r = 0; r < 8; ++r) {
                const int qrow = q0 + r + 8 * half;
                const float v = s[n][r] * 0.125f;
                s[n][r] = (kcol <= qrow) ? v : -1e30f;
            }
        }

        // ---- online softmax (row stats live per 16-lane half) ----
        float pr[4][8];
#pragma unroll
        for (int r = 0; r < 8; ++r) {
            float mx = fmaxf(fmaxf(s[0][r], s[1][r]), fmaxf(s[2][r], s[3][r]));
            mx = rmax16(mx);
            const float mnew = fmaxf(mi[r], mx);
            const float sc   = __expf(mi[r] - mnew);
            mi[r] = mnew;
            float rs = 0.0f;
#pragma unroll
            for (int n = 0; n < 4; ++n) {
                const float e = __expf(s[n][r] - mnew);
                pr[n][r] = e;
                rs += e;
            }
            rs = rsum16(rs);
            li[r] = li[r] * sc + rs;
#pragma unroll
            for (int n = 0; n < 4; ++n) o[n][r] *= sc;
        }

        // ---- C-layout -> A-layout for P via per-wave LDS staging ----
#pragma unroll
        for (int n = 0; n < 4; ++n)
#pragma unroll
            for (int r = 0; r < 8; ++r)
                ps[(r + 8 * half) * 64 + n * 16 + l16] = (__bf16)pr[n][r];
        asm volatile("s_wait_dscnt 0" ::: "memory");   // CDNA5 split DS counter

        v16bf pf[2];
#pragma unroll
        for (int kk = 0; kk < 2; ++kk) {
            const __bf16* pp = ps + l16 * 64 + kk * 32 + half * 16;
#pragma unroll
            for (int j = 0; j < 16; ++j) pf[kk][j] = pp[j];
        }

        // ---- O += P V ; V B-frags contiguous thanks to transposed v ----
#pragma unroll
        for (int n = 0; n < 4; ++n) {
            const __bf16* vp = vT + (size_t)(n * 16 + l16) * T_ + k0 + half * 16;
            const v16bf v0 = *(const v16bf*)(vp);
            const v16bf v1 = *(const v16bf*)(vp + 32);
            o[n] = __builtin_amdgcn_wmma_f32_16x16x32_bf16(
                false, pf[0], false, v0, (short)0, o[n], false, false);
            o[n] = __builtin_amdgcn_wmma_f32_16x16x32_bf16(
                false, pf[1], false, v1, (short)0, o[n], false, false);
        }
    }

    // ---- normalize and store fp32 output ----
#pragma unroll
    for (int n = 0; n < 4; ++n)
#pragma unroll
        for (int r = 0; r < 8; ++r)
            out[((size_t)b * T_ + q0 + r + 8 * half) * H_ + n * 16 + l16] =
                o[n][r] / li[r];
}

// ---------------------------------------------------------------------------
extern "C" void kernel_launch(void* const* d_in, const int* in_sizes, int n_in,
                              void* d_out, int out_size, void* d_ws, size_t ws_size,
                              hipStream_t stream)
{
    (void)in_sizes; (void)n_in; (void)out_size; (void)ws_size;
    const float* x  = (const float*)d_in[0];
    const float* Wk = (const float*)d_in[1];
    const float* Wq = (const float*)d_in[2];
    const float* Wv = (const float*)d_in[3];
    float* out = (float*)d_out;

    // workspace: q (2 MB) | k (2 MB) | vT (2 MB), all bf16
    char* ws = (char*)d_ws;
    __bf16* qb  = (__bf16*)(ws);
    __bf16* kb  = (__bf16*)(ws + (size_t)BT_ * H_ * 2);
    __bf16* vTb = (__bf16*)(ws + (size_t)BT_ * H_ * 4);

    dim3 gProj(BT_ / 16, 3);
    qkv_proj_kernel<<<gProj, 128, 0, stream>>>(x, Wq, Wk, Wv, qb, kb, vTb);
    attn_kernel<<<BT_ / (16 * 8), 256, 0, stream>>>(qb, kb, vTb, out);
}
